// LightweightGNN_3229815407293
// MI455X (gfx1250) — matrix operand order repaired and checked by reading, not compiled, not run on hardware
//
#include <hip/hip_runtime.h>
#include <hip/hip_bf16.h>

typedef __attribute__((ext_vector_type(16))) _Float16 v16h;
typedef __attribute__((ext_vector_type(8)))  float    v8f;

#define IN_CH 64
#define HID   128

// ---------------------------------------------------------------------------
// Kernel 1: zero the aggregation workspace (N * 64 floats)
// ---------------------------------------------------------------------------
__global__ void gnn_zero_kernel(float4* __restrict__ ws, long long n_vec4) {
    long long i = (long long)blockIdx.x * blockDim.x + threadIdx.x;
    long long stride = (long long)gridDim.x * blockDim.x;
    float4 z = {0.f, 0.f, 0.f, 0.f};
    for (; i < n_vec4; i += stride) ws[i] = z;
}

// ---------------------------------------------------------------------------
// Kernel 2: edge scatter-add  aggr[row[e]][c] += x[col[e]][c]
// 4 edges per 256-thread block, one lane per channel (coalesced 256B rows).
// unsafeAtomicAdd -> global_atomic_add_f32 (no-return, STOREcnt path, L2 RMW).
// ---------------------------------------------------------------------------
__global__ void gnn_scatter_kernel(const float* __restrict__ x,
                                   const int* __restrict__ row,
                                   const int* __restrict__ col,
                                   float* __restrict__ aggr,
                                   int E) {
    int e = blockIdx.x * 4 + (threadIdx.x >> 6);
    int c = threadIdx.x & 63;
    if (e < E) {
        int r = row[e];
        int s = col[e];
        float v = x[(long long)s * IN_CH + c];
        unsafeAtomicAdd(&aggr[(long long)r * IN_CH + c], v);
    }
}

// ---------------------------------------------------------------------------
// Kernel 3: fused MLP  scores = relu(aggr @ W1 + b1) @ W2 + b2
// 8 waves/block, 16 nodes/wave (128 nodes/block).
// GEMM1 via v_wmma_f32_16x16x32_f16, K=64 in two K=32 steps, 8 N-tiles of 16.
// GEMM2 fused on the f32 accumulator layout + shfl_xor reduction.
// ---------------------------------------------------------------------------
__global__ void __launch_bounds__(256)
gnn_mlp_kernel(const float* __restrict__ aggr,
               const float* __restrict__ W1,  // [64][128] row-major
               const float* __restrict__ b1,  // [128]
               const float* __restrict__ W2,  // [128]
               const float* __restrict__ b2,  // [1]
               float* __restrict__ out,       // [N]
               int N) {
    // W1 pre-swizzled into B-fragment order:
    // frag f = t*2 + s (tile t in 0..7, k-step s in 0..1), lane L, j in 0..15:
    //   sW1[((f*32 + L)*16) + j] = W1[ s*32 + (L>>4)*16 + j ][ t*16 + (L&15) ]
    // so each lane's v16h B-fragment is one contiguous 32B LDS read.
    __shared__ __attribute__((aligned(32))) _Float16 sW1[16 * 32 * 16]; // 16 KB
    __shared__ float sB1[HID];
    __shared__ float sW2[HID];

    int tid = threadIdx.x;
    for (int i = tid; i < 16 * 32 * 16; i += 256) {
        int j    = i & 15;
        int lane = (i >> 4) & 31;
        int f    = i >> 9;          // 0..15
        int t    = f >> 1;
        int s    = f & 1;
        int k = s * 32 + ((lane >> 4) << 4) + j;
        int n = t * 16 + (lane & 15);
        sW1[i] = (_Float16)W1[k * HID + n];
    }
    if (tid < HID) {
        sB1[tid] = b1[tid];
        sW2[tid] = W2[tid];
    }
    __syncthreads();

    const float bias2 = b2[0];

    int wave = tid >> 5;
    int lane = tid & 31;
    int half = lane >> 4;       // K-group selector for A, M-group for C/D
    int mcol = lane & 15;

    int node_base = blockIdx.x * 128 + wave * 16;
    if (node_base >= N) return;

    // ---- A fragments: 16x32 f16, lane L = row (L&15), K groups by half ----
    int m = node_base + mcol;
    if (m >= N) m = N - 1;                    // clamp (EXEC must stay all-1s)
    const float* arow = aggr + (long long)m * IN_CH;

    v16h afrag[2];
#pragma unroll
    for (int s = 0; s < 2; ++s) {
        int b0 = s * 32 + half * 16;
#pragma unroll
        for (int j = 0; j < 8; ++j) {
            afrag[s][j]     = (_Float16)arow[b0 + j];        // K = b0 + j
            afrag[s][8 + j] = (_Float16)arow[b0 + 8 + j];    // K = b0 + 8 + j
        }
    }

    // ---- tile loop: 8 N-tiles of 16 columns of HID ----
    float score[8];
#pragma unroll
    for (int r = 0; r < 8; ++r) score[r] = 0.f;

#pragma unroll
    for (int t = 0; t < 8; ++t) {
        const v16h bf0 = *(const v16h*)(sW1 + ((t * 2 + 0) * 32 + lane) * 16);
        const v16h bf1 = *(const v16h*)(sW1 + ((t * 2 + 1) * 32 + lane) * 16);

        v8f acc = {};
        acc = __builtin_amdgcn_wmma_f32_16x16x32_f16(
                  false, afrag[0], false, bf0, (short)0, acc, false, false);
        acc = __builtin_amdgcn_wmma_f32_16x16x32_f16(
                  false, afrag[1], false, bf1, (short)0, acc, false, false);

        // h = relu(acc + b1[n]); partial score += h * W2[n]
        int n = t * 16 + mcol;
        float b1v = sB1[n];
        float w2v = sW2[n];
#pragma unroll
        for (int r = 0; r < 8; ++r) {
            float hv = acc[r] + b1v;
            hv = hv > 0.f ? hv : 0.f;
            score[r] = __builtin_fmaf(hv, w2v, score[r]);
        }
    }

    // ---- reduce over the 16 lanes of each half (N dimension) ----
#pragma unroll
    for (int r = 0; r < 8; ++r) {
        float v = score[r];
        v += __shfl_xor(v, 8, 32);
        v += __shfl_xor(v, 4, 32);
        v += __shfl_xor(v, 2, 32);
        v += __shfl_xor(v, 1, 32);
        score[r] = v;
    }

    // C/D layout: half 0 -> VGPR r holds M=r; half 1 -> M=8+r
    if (mcol == 0) {
        int mbase = node_base + half * 8;
#pragma unroll
        for (int r = 0; r < 8; ++r) {
            int mm = mbase + r;
            if (mm < N) out[mm] = score[r] + bias2;
        }
    }
}

// ---------------------------------------------------------------------------
extern "C" void kernel_launch(void* const* d_in, const int* in_sizes, int n_in,
                              void* d_out, int out_size, void* d_ws, size_t ws_size,
                              hipStream_t stream) {
    const float* x   = (const float*)d_in[0];          // [N, 64]
    const int*   ei  = (const int*)d_in[1];            // [2, E] (int32)
    const float* W1  = (const float*)d_in[2];          // [64, 128]
    const float* b1  = (const float*)d_in[3];          // [128]
    const float* W2  = (const float*)d_in[4];          // [128, 1]
    const float* b2  = (const float*)d_in[5];          // [1]
    float*       out = (float*)d_out;                  // [N]

    int N = in_sizes[0] / IN_CH;
    int E = in_sizes[1] / 2;
    const int* row = ei;
    const int* col = ei + E;

    float* aggr = (float*)d_ws;                        // N * 64 floats

    // 1) zero aggr
    long long n_vec4 = (long long)N * IN_CH / 4;
    int zblocks = (int)((n_vec4 + 255) / 256);
    if (zblocks > 4096) zblocks = 4096;
    gnn_zero_kernel<<<zblocks, 256, 0, stream>>>((float4*)aggr, n_vec4);

    // 2) edge scatter-add (4 edges / block)
    int sblocks = (E + 3) / 4;
    gnn_scatter_kernel<<<sblocks, 256, 0, stream>>>(x, row, col, aggr, E);

    // 3) fused WMMA MLP (128 nodes / block)
    int mblocks = (N + 127) / 128;
    gnn_mlp_kernel<<<mblocks, 256, 0, stream>>>(aggr, W1, b1, W2, b2, out, N);
}
